// WeightedPolynormerGlobal_59596966199879
// MI455X (gfx1250) — matrix operand order repaired and checked by reading, not compiled, not run on hardware
//
#include <hip/hip_runtime.h>

#define DMODEL 256
#define HEADS  8
#define DHEAD  32
#define BATCH  8
#define NMAX   8192
#define MROWS  (BATCH * NMAX)   // 65536
#define LNEPS  1e-5f

typedef __attribute__((ext_vector_type(16))) __bf16 v16bf;
typedef __attribute__((ext_vector_type(8)))  float  v8f;

union FragB16 { uint4 u[2]; v16bf v; };
union Pack8   { unsigned short s[8]; uint4 u; };

__device__ __forceinline__ unsigned short f2bf(float f) {
  unsigned int u = __float_as_uint(f);
  u += 0x7FFFu + ((u >> 16) & 1u);           // round-to-nearest-even
  return (unsigned short)(u >> 16);
}

// ---------------------------------------------------------------------------
// Kernel 1: LayerNorm(x) -> x0 (f32) and x0 (bf16).  One wave32 per 256-row.
// ---------------------------------------------------------------------------
__global__ void __launch_bounds__(256) ln1_kernel(
    const float* __restrict__ x, const float* __restrict__ g,
    const float* __restrict__ b, float* __restrict__ x0,
    unsigned short* __restrict__ x0b) {
  int warp = threadIdx.x >> 5, lane = threadIdx.x & 31;
  size_t row = (size_t)blockIdx.x * 8 + warp;
  const float* xr = x + row * DMODEL + lane * 8;
  float4 a0 = *(const float4*)xr;
  float4 a1 = *(const float4*)(xr + 4);
  float s  = a0.x + a0.y + a0.z + a0.w + a1.x + a1.y + a1.z + a1.w;
  float sq = a0.x*a0.x + a0.y*a0.y + a0.z*a0.z + a0.w*a0.w
           + a1.x*a1.x + a1.y*a1.y + a1.z*a1.z + a1.w*a1.w;
#pragma unroll
  for (int off = 16; off > 0; off >>= 1) {
    s  += __shfl_xor(s,  off, 32);
    sq += __shfl_xor(sq, off, 32);
  }
  float mu  = s * (1.0f / DMODEL);
  float var = sq * (1.0f / DMODEL) - mu * mu;
  float inv = rsqrtf(var + LNEPS);
  int c0 = lane * 8;
  float4 g0 = *(const float4*)(g + c0), g1 = *(const float4*)(g + c0 + 4);
  float4 b0 = *(const float4*)(b + c0), b1 = *(const float4*)(b + c0 + 4);
  float o[8];
  o[0] = (a0.x - mu) * inv * g0.x + b0.x; o[1] = (a0.y - mu) * inv * g0.y + b0.y;
  o[2] = (a0.z - mu) * inv * g0.z + b0.z; o[3] = (a0.w - mu) * inv * g0.w + b0.w;
  o[4] = (a1.x - mu) * inv * g1.x + b1.x; o[5] = (a1.y - mu) * inv * g1.y + b1.y;
  o[6] = (a1.z - mu) * inv * g1.z + b1.z; o[7] = (a1.w - mu) * inv * g1.w + b1.w;
  float* x0r = x0 + row * DMODEL + c0;
  *(float4*)x0r       = make_float4(o[0], o[1], o[2], o[3]);
  *(float4*)(x0r + 4) = make_float4(o[4], o[5], o[6], o[7]);
  Pack8 p;
#pragma unroll
  for (int j = 0; j < 8; ++j) p.s[j] = f2bf(o[j]);
  *(uint4*)(x0b + row * DMODEL + c0) = p.u;
}

// ---------------------------------------------------------------------------
// Kernel 2: W (DxD f32, K-major) -> W^T (N-major, bf16)
// ---------------------------------------------------------------------------
__global__ void __launch_bounds__(256) wt_kernel(
    const float* __restrict__ W, unsigned short* __restrict__ Wt) {
  int f = blockIdx.x * 256 + threadIdx.x;   // 0..65535
  int n = f >> 8, k = f & 255;
  Wt[n * DMODEL + k] = f2bf(W[k * DMODEL + n]);
}

// ---------------------------------------------------------------------------
// Kernel 3: bf16 WMMA GEMM, C[M,256] = A[M,256] @ W + epilogue.
// Block tile 128x256, 8 waves, wave w owns rows [16w,16w+16) x 16 N-tiles.
// ---------------------------------------------------------------------------
enum { EPI_H = 0, EPI_SIG = 1, EPI_SIGP = 2, EPI_ADD = 3, EPI_OUT = 4 };

template <int EPI>
__global__ void __launch_bounds__(256) gemm_kernel(
    const unsigned short* __restrict__ A,   // [M,256] bf16 row-major
    const unsigned short* __restrict__ Bt,  // [256,256] bf16, [n][k]
    const float* __restrict__ bias,         // [256]
    const float* __restrict__ aux,          // per-EPI extra operand
    float* __restrict__ out) {
  __shared__ alignas(16) unsigned short Atile[128 * 32];  //  8 KB
  __shared__ alignas(16) unsigned short Btile[256 * 32];  // 16 KB
  const int tid = threadIdx.x;
  const int wv = tid >> 5, lane = tid & 31;
  const int half = lane >> 4, l16 = lane & 15;
  const size_t rowBase = (size_t)blockIdx.x * 128;

  v8f acc[16];
#pragma unroll
  for (int i = 0; i < 16; ++i)
#pragma unroll
    for (int j = 0; j < 8; ++j) acc[i][j] = 0.0f;

  for (int kk = 0; kk < 8; ++kk) {
    // Stage A tile 128x32 and B tile 256x32 (bf16) via 16B copies.
#pragma unroll
    for (int i = 0; i < 2; ++i) {
      int f = tid * 2 + i;                  // 0..511
      int r = f >> 2, seg = f & 3;
      *(uint4*)&Atile[r * 32 + seg * 8] =
          *(const uint4*)&A[(rowBase + r) * DMODEL + kk * 32 + seg * 8];
    }
#pragma unroll
    for (int j = 0; j < 4; ++j) {
      int f = tid + j * 256;                // 0..1023
      int n = f >> 2, seg = f & 3;
      *(uint4*)&Btile[n * 32 + seg * 8] =
          *(const uint4*)&Bt[n * DMODEL + kk * 32 + seg * 8];
    }
    __syncthreads();
    // A fragment (16x32): lane l -> M = l&15, halves of K split by l>>4.
    FragB16 af;
    af.u[0] = *(const uint4*)&Atile[(wv * 16 + l16) * 32 + half * 8];
    af.u[1] = *(const uint4*)&Atile[(wv * 16 + l16) * 32 + 16 + half * 8];
#pragma unroll
    for (int nt = 0; nt < 16; ++nt) {
      // B fragment (32x16): lane l -> N = nt*16 + (l&15), K = (l>>4)*8 .. +7 and 16 + (l>>4)*8 .. +7.
      FragB16 bfr;
      bfr.u[0] = *(const uint4*)&Btile[(nt * 16 + l16) * 32 + half * 8];
      bfr.u[1] = *(const uint4*)&Btile[(nt * 16 + l16) * 32 + 16 + half * 8];
      acc[nt] = __builtin_amdgcn_wmma_f32_16x16x32_bf16(
          false, af.v, false, bfr.v, (short)0, acc[nt], false, false);
    }
    __syncthreads();
  }
  // Epilogue: C layout -> VGPR r: M = r + 8*(l>>4), N = nt*16 + (l&15).
#pragma unroll
  for (int nt = 0; nt < 16; ++nt) {
    int n = nt * 16 + l16;
    float bn = bias[n];
#pragma unroll
    for (int r = 0; r < 8; ++r) {
      size_t gm = rowBase + wv * 16 + r + 8 * half;
      float v = acc[nt][r] + bn;
      if (EPI == EPI_H)        v = v + aux[n];                         // + beta
      else if (EPI == EPI_SIG) v = 1.0f / (1.0f + __expf(-v));
      else if (EPI == EPI_SIGP) v = (1.0f / (1.0f + __expf(-v))) * aux[gm];
      else if (EPI == EPI_OUT) v = fmaxf(v, 0.0f) + aux[gm * DMODEL + n];
      out[gm * DMODEL + n] = v;
    }
  }
}

// ---------------------------------------------------------------------------
// Kernel 4: kv[b,h,d,e] = sum_n k*v ; ksum[b,h,d] = sum_n k  (f32 atomics)
// grid = (64 bh, 16 n-chunks), 256 threads; thread owns (d, 4 e's).
// ---------------------------------------------------------------------------
__global__ void __launch_bounds__(256) kv_kernel(
    const float* __restrict__ kf, const float* __restrict__ vf,
    float* __restrict__ kv, float* __restrict__ ksum) {
  __shared__ alignas(16) float kc[32 * 32];
  __shared__ alignas(16) float vc[32 * 32];
  const int bh = blockIdx.x;                // 0..63
  const int b = bh >> 3, h = bh & 7;
  const int tid = threadIdx.x;
  const int d = tid >> 3, eg = (tid & 7) * 4;
  float a0 = 0, a1 = 0, a2 = 0, a3 = 0, ks = 0;
  const size_t base = (size_t)b * NMAX + (size_t)blockIdx.y * 512;
  for (int c = 0; c < 512; c += 32) {
    int nl = tid >> 3, f4 = (tid & 7) * 4;
    size_t row = base + c + nl;
    *(float4*)&kc[nl * 32 + f4] = *(const float4*)&kf[row * DMODEL + h * 32 + f4];
    *(float4*)&vc[nl * 32 + f4] = *(const float4*)&vf[row * DMODEL + h * 32 + f4];
    __syncthreads();
#pragma unroll 8
    for (int nn = 0; nn < 32; ++nn) {
      float kd = kc[nn * 32 + d];
      float4 vv = *(const float4*)&vc[nn * 32 + eg];
      a0 += kd * vv.x; a1 += kd * vv.y; a2 += kd * vv.z; a3 += kd * vv.w;
      ks += kd;
    }
    __syncthreads();
  }
  float* kvp = kv + bh * 1024 + d * 32 + eg;
  atomicAdd(kvp + 0, a0); atomicAdd(kvp + 1, a1);
  atomicAdd(kvp + 2, a2); atomicAdd(kvp + 3, a3);
  if ((tid & 7) == 0) atomicAdd(ksum + bh * 32 + d, ks);
}

// ---------------------------------------------------------------------------
// Kernel 5: att = (q . kv)/(q . ksum); y = LN2(att) * hpb  -> bf16
// One wave32 per row; kv[b] + q row staged in LDS.
// ---------------------------------------------------------------------------
__global__ void __launch_bounds__(256) attn_kernel(
    const float* __restrict__ qf, const float* __restrict__ kv,
    const float* __restrict__ ksum, const float* __restrict__ hpb,
    const float* __restrict__ g2, const float* __restrict__ b2,
    unsigned short* __restrict__ yb) {
  __shared__ alignas(16) float kvs[HEADS * 32 * 32];  // 32 KB
  __shared__ float kss[HEADS * 32];
  __shared__ alignas(16) float qs[8 * DMODEL];        //  8 KB
  const int tid = threadIdx.x;
  const int warp = tid >> 5, lane = tid & 31;
  const size_t row = (size_t)blockIdx.x * 8 + warp;
  const int b = (int)(row >> 13);                     // 8192 rows per batch
#pragma unroll
  for (int i = 0; i < 8; ++i) {
    int f = tid + i * 256;                            // float4 index 0..2047
    *(float4*)&kvs[f * 4] = *(const float4*)&kv[(size_t)b * 8192 + f * 4];
  }
  kss[tid & 255] = ksum[b * 256 + (tid & 255)];
  const int c0 = lane * 8;
  const float* qr = qf + row * DMODEL + c0;
  *(float4*)&qs[warp * DMODEL + c0]     = *(const float4*)qr;
  *(float4*)&qs[warp * DMODEL + c0 + 4] = *(const float4*)(qr + 4);
  __syncthreads();

  const int h = lane >> 2;
  const int e0 = (lane & 3) * 8;        // lane's global cols = h*32+e0.. = c0..
  float4 n0 = make_float4(0, 0, 0, 0), n1 = make_float4(0, 0, 0, 0);
  float den = 0.0f;
  const float* qh  = &qs[warp * DMODEL + h * 32];
  const float* kvh = &kvs[h * 1024];
  const float* ksh = &kss[h * 32];
#pragma unroll 8
  for (int dd = 0; dd < 32; ++dd) {
    float qd = qh[dd];
    float4 k0 = *(const float4*)&kvh[dd * 32 + e0];
    float4 k1 = *(const float4*)&kvh[dd * 32 + e0 + 4];
    n0.x += qd * k0.x; n0.y += qd * k0.y; n0.z += qd * k0.z; n0.w += qd * k0.w;
    n1.x += qd * k1.x; n1.y += qd * k1.y; n1.z += qd * k1.z; n1.w += qd * k1.w;
    den += qd * ksh[dd];
  }
  float id = 1.0f / den;
  float att[8] = {n0.x * id, n0.y * id, n0.z * id, n0.w * id,
                  n1.x * id, n1.y * id, n1.z * id, n1.w * id};
  float s = 0, sq = 0;
#pragma unroll
  for (int j = 0; j < 8; ++j) { s += att[j]; sq += att[j] * att[j]; }
#pragma unroll
  for (int off = 16; off > 0; off >>= 1) {
    s  += __shfl_xor(s,  off, 32);
    sq += __shfl_xor(sq, off, 32);
  }
  float mu  = s * (1.0f / DMODEL);
  float var = sq * (1.0f / DMODEL) - mu * mu;
  float inv = rsqrtf(var + LNEPS);
  float4 gA = *(const float4*)(g2 + c0), gB = *(const float4*)(g2 + c0 + 4);
  float4 bA = *(const float4*)(b2 + c0), bB = *(const float4*)(b2 + c0 + 4);
  const float* hr = hpb + row * DMODEL + c0;
  float4 hA = *(const float4*)hr, hB = *(const float4*)(hr + 4);
  float gg[8] = {gA.x, gA.y, gA.z, gA.w, gB.x, gB.y, gB.z, gB.w};
  float bb[8] = {bA.x, bA.y, bA.z, bA.w, bB.x, bB.y, bB.z, bB.w};
  float hh[8] = {hA.x, hA.y, hA.z, hA.w, hB.x, hB.y, hB.z, hB.w};
  Pack8 p;
#pragma unroll
  for (int j = 0; j < 8; ++j)
    p.s[j] = f2bf(((att[j] - mu) * inv * gg[j] + bb[j]) * hh[j]);
  *(uint4*)(yb + row * DMODEL + c0) = p.u;
}

// ---------------------------------------------------------------------------
extern "C" void kernel_launch(void* const* d_in, const int* in_sizes, int n_in,
                              void* d_out, int out_size, void* d_ws,
                              size_t ws_size, hipStream_t stream) {
  (void)in_sizes; (void)n_in; (void)out_size; (void)ws_size;
  const float* x        = (const float*)d_in[0];
  const float* nodeProb = (const float*)d_in[1];
  const float* Wh   = (const float*)d_in[2];  const float* bh   = (const float*)d_in[3];
  const float* Wq   = (const float*)d_in[4];  const float* bq   = (const float*)d_in[5];
  const float* Wk   = (const float*)d_in[6];  const float* bk   = (const float*)d_in[7];
  const float* Wv   = (const float*)d_in[8];  const float* bv   = (const float*)d_in[9];
  const float* Wout = (const float*)d_in[10]; const float* bout = (const float*)d_in[11];
  const float* ln1g = (const float*)d_in[12]; const float* ln1b = (const float*)d_in[13];
  const float* ln2g = (const float*)d_in[14]; const float* ln2b = (const float*)d_in[15];
  const float* betab = (const float*)d_in[16];

  char* ws = (char*)d_ws;
  size_t off = 0;
  auto carve = [&](size_t bytes) {
    void* p = ws + off;
    off = (off + bytes + 255) & ~(size_t)255;
    return p;
  };
  const size_t MD4 = (size_t)MROWS * DMODEL * sizeof(float);      // 64 MB
  const size_t MD2 = (size_t)MROWS * DMODEL * sizeof(short);      // 32 MB
  float* x0f  = (float*)carve(MD4);
  float* hpb  = (float*)carve(MD4);
  float* qf   = (float*)carve(MD4);
  float* kf   = (float*)carve(MD4);
  float* vf   = (float*)carve(MD4);
  unsigned short* x0b = (unsigned short*)carve(MD2);
  unsigned short* yb  = (unsigned short*)carve(MD2);
  unsigned short* WhT = (unsigned short*)carve(DMODEL * DMODEL * 2);
  unsigned short* WqT = (unsigned short*)carve(DMODEL * DMODEL * 2);
  unsigned short* WkT = (unsigned short*)carve(DMODEL * DMODEL * 2);
  unsigned short* WvT = (unsigned short*)carve(DMODEL * DMODEL * 2);
  unsigned short* WoT = (unsigned short*)carve(DMODEL * DMODEL * 2);
  float* kv   = (float*)carve(64 * 1024 * sizeof(float));         // 256 KB
  float* ksum = (float*)carve(64 * 32 * sizeof(float));           //   8 KB

  // 1) LN1
  ln1_kernel<<<MROWS / 8, 256, 0, stream>>>(x, ln1g, ln1b, x0f, x0b);
  // 2) weight convert/transpose to bf16
  wt_kernel<<<256, 256, 0, stream>>>(Wh, WhT);
  wt_kernel<<<256, 256, 0, stream>>>(Wq, WqT);
  wt_kernel<<<256, 256, 0, stream>>>(Wk, WkT);
  wt_kernel<<<256, 256, 0, stream>>>(Wv, WvT);
  wt_kernel<<<256, 256, 0, stream>>>(Wout, WoT);
  // 3) four projection GEMMs with fused epilogues
  const int GB = MROWS / 128;  // 512 blocks
  gemm_kernel<EPI_H><<<GB, 256, 0, stream>>>(x0b, WhT, bh, betab, hpb);
  gemm_kernel<EPI_SIG><<<GB, 256, 0, stream>>>(x0b, WqT, bq, nullptr, qf);
  gemm_kernel<EPI_SIGP><<<GB, 256, 0, stream>>>(x0b, WkT, bk, nodeProb, kf);
  gemm_kernel<EPI_ADD><<<GB, 256, 0, stream>>>(x0b, WvT, bv, nullptr, vf);
  // 4) kv / ksum reduction (zero accumulators first; kv and ksum adjacent)
  hipMemsetAsync(kv, 0, 64 * 1024 * sizeof(float) + 256 /*pad*/ +
                           64 * 32 * sizeof(float), stream);
  kv_kernel<<<dim3(64, 16), 256, 0, stream>>>(kf, vf, kv, ksum);
  // 5) attention mix + LN2 * (h + beta) -> bf16
  attn_kernel<<<MROWS / 8, 256, 0, stream>>>(qf, kv, ksum, hpb, ln2g, ln2b, yb);
  // 6) output GEMM with fused relu + residual(x0)
  gemm_kernel<EPI_OUT><<<GB, 256, 0, stream>>>(yb, WoT, bout, x0f, (float*)d_out);
}
